// ABPTModelB_13486197310045
// MI455X (gfx1250) — matrix operand order repaired and checked by reading, not compiled, not run on hardware
//
#include <hip/hip_runtime.h>
#include <cstddef>
#include <cstdint>

// ---------------- constants (match reference) ----------------
#define BB 4
#define TT 1024
#define DD 768
#define HH 12
#define HDIM 64
#define LL 4
#define VV 32000
#define FFD 3072
#define PHD 192
#define BT (BB*TT)          // 4096 rows

// ---------------- types ----------------
typedef __bf16 bf16x16 __attribute__((ext_vector_type(16)));
typedef float  f32x8   __attribute__((ext_vector_type(8)));

// hardware f32 -> bf16 (v_cvt_pk_bf16_f32 on gfx1250)
static __device__ inline __bf16 f2bf(float f) { return (__bf16)f; }

static __device__ inline f32x8 wmma_bf16(bf16x16 a, bf16x16 b, f32x8 c) {
  return __builtin_amdgcn_wmma_f32_16x16x32_bf16(false, a, false, b, (short)0, c, false, false);
}

static __device__ inline float gelu_f(float x) {
  const float k = 0.7978845608028654f;           // sqrt(2/pi)
  return 0.5f * x * (1.0f + tanhf(k * (x + 0.044715f * x * x * x)));
}

// load a 16-bit A fragment (16x32 tile, lane = row) from row pointer at (k0 + half*16)
static __device__ inline bf16x16 load_afrag(const float* p) {
  float4 a0 = *(const float4*)(p);
  float4 a1 = *(const float4*)(p + 4);
  float4 a2 = *(const float4*)(p + 8);
  float4 a3 = *(const float4*)(p + 12);
  bf16x16 af;
  af[0] = f2bf(a0.x); af[1] = f2bf(a0.y); af[2]  = f2bf(a0.z); af[3]  = f2bf(a0.w);
  af[4] = f2bf(a1.x); af[5] = f2bf(a1.y); af[6]  = f2bf(a1.z); af[7]  = f2bf(a1.w);
  af[8] = f2bf(a2.x); af[9] = f2bf(a2.y); af[10] = f2bf(a2.z); af[11] = f2bf(a2.w);
  af[12] = f2bf(a3.x); af[13] = f2bf(a3.y); af[14] = f2bf(a3.z); af[15] = f2bf(a3.w);
  return af;
}

// ---------------- embedding ----------------
__global__ __launch_bounds__(256)
void embed_kernel(const int* __restrict__ ids, const float* __restrict__ te,
                  const float* __restrict__ pe, float* __restrict__ x) {
  int i = blockIdx.x * 256 + threadIdx.x;
  if (i >= BT * DD) return;
  int d  = i % DD;
  int bt = i / DD;
  int t  = bt % TT;
  int id = ids[bt];
  x[i] = te[(size_t)id * DD + d] + pe[(size_t)t * DD + d];
}

// ---------------- layernorm (one block per row, D=768) ----------------
__global__ __launch_bounds__(256)
void ln_kernel(const float* __restrict__ x, const float* __restrict__ g,
               const float* __restrict__ bb, float* __restrict__ out) {
  __shared__ float red[256];
  const int row = blockIdx.x, tid = threadIdx.x;
  const float* xr = x + (size_t)row * DD;
  float v0 = xr[tid], v1 = xr[tid + 256], v2 = xr[tid + 512];
  red[tid] = v0 + v1 + v2;
  __syncthreads();
  for (int o = 128; o > 0; o >>= 1) { if (tid < o) red[tid] += red[tid + o]; __syncthreads(); }
  const float mean = red[0] * (1.0f / DD);
  __syncthreads();
  float d0 = v0 - mean, d1 = v1 - mean, d2 = v2 - mean;
  red[tid] = d0 * d0 + d1 * d1 + d2 * d2;
  __syncthreads();
  for (int o = 128; o > 0; o >>= 1) { if (tid < o) red[tid] += red[tid + o]; __syncthreads(); }
  const float rstd = rsqrtf(red[0] * (1.0f / DD) + 1e-5f);
  float* orow = out + (size_t)row * DD;
  orow[tid]       = d0 * rstd * g[tid]       + bb[tid];
  orow[tid + 256] = d1 * rstd * g[tid + 256] + bb[tid + 256];
  orow[tid + 512] = d2 * rstd * g[tid + 512] + bb[tid + 512];
}

// ---------------- ed = mean(x*x, -1) ----------------
__global__ __launch_bounds__(256)
void rowmeansq_kernel(const float* __restrict__ x, float* __restrict__ ed) {
  __shared__ float red[256];
  const int row = blockIdx.x, tid = threadIdx.x;
  const float* xr = x + (size_t)row * DD;
  float v0 = xr[tid], v1 = xr[tid + 256], v2 = xr[tid + 512];
  red[tid] = v0 * v0 + v1 * v1 + v2 * v2;
  __syncthreads();
  for (int o = 128; o > 0; o >>= 1) { if (tid < o) red[tid] += red[tid + o]; __syncthreads(); }
  if (tid == 0) ed[row] = red[0] * (1.0f / DD);
}

// ---------------- quantile thresholds via single-block bitonic sort ----------------
__global__ __launch_bounds__(256)
void quantile_kernel(const float* __restrict__ ed, float* __restrict__ th) {
  __shared__ float s[BT];
  const int tid = threadIdx.x;
  for (int i = tid; i < BT; i += 256) s[i] = ed[i];
  __syncthreads();
  for (int k = 2; k <= BT; k <<= 1) {
    for (int j = k >> 1; j > 0; j >>= 1) {
      for (int i = tid; i < BT; i += 256) {
        int ixj = i ^ j;
        if (ixj > i) {
          bool up = (i & k) == 0;
          float a = s[i], b = s[ixj];
          if ((a > b) == up) { s[i] = b; s[ixj] = a; }
        }
      }
      __syncthreads();
    }
  }
  if (tid < 3) {
    const float q[3] = {0.7f, 0.8f, 0.9f};
    float pos = q[tid] * (float)(BT - 1);
    int i0 = (int)pos;
    float fr = pos - (float)i0;
    int i1 = i0 + 1 < BT ? i0 + 1 : BT - 1;
    th[tid] = s[i0] + (s[i1] - s[i0]) * fr;
  }
}

// ---------------- conditional select x[i] = (route==target) ? cand[i] : x[i] ----------------
__global__ __launch_bounds__(256)
void select_kernel(float* __restrict__ x, const float* __restrict__ cand,
                   const float* __restrict__ ed, const float* __restrict__ th, int target) {
  int i = blockIdx.x * 256 + threadIdx.x;
  if (i >= BT * DD) return;
  float e = ed[i / DD];
  int route = (e > th[0]) + (e > th[1]) + (e > th[2]);
  if (route == target) x[i] = cand[i];
}

// ---------------- WMMA GEMM: C[M,N] = epilogue(A[M,K] @ B[K,N]) ----------------
// block = 128 threads (4 waves). Block tile 128(M) x 64(N); each wave owns two
// 16-row M-subtiles (at m0 and m0+64) over the full 64-col N tile.
// The 32x64 B tile is staged in LDS ALREADY PERMUTED into WMMA B-fragment layout:
//   lbp[t][lane][e] = bf16( B[k0 + 16*(lane/16) + e][n0 + 16*t + lane%16] )
// so each wave fetches a whole v16bf fragment with two ds_load_b128.
// EP: 0 = none, 1 = C = Res + A@B, 2 = C = gelu(A@B)
template<int EP>
__global__ __launch_bounds__(128)
void gemm_wmma(const float* __restrict__ A, const float* __restrict__ Bm,
               const float* __restrict__ Res, float* __restrict__ C,
               int K, int lda, int ldb, int ldc) {
  __shared__ alignas(32) __bf16 lbp[4][32][16];
  const int tid  = threadIdx.x;
  const int wave = tid >> 5;
  const int lane = tid & 31;
  const int half = lane >> 4;
  const int lm   = lane & 15;
  const int n0   = blockIdx.x * 64;
  const int m0   = blockIdx.y * 128 + wave * 16;

  // writer mapping: thread -> (column nn in [0,64), kk-halve wkh in {0,1})
  const int wnn   = tid & 63;
  const int wkh   = tid >> 6;
  const int wt    = wnn >> 4;
  const int wlane = (wnn & 15) + 16 * wkh;

  f32x8 acc[2][4];
#pragma unroll
  for (int s = 0; s < 2; ++s)
#pragma unroll
    for (int t = 0; t < 4; ++t)
#pragma unroll
      for (int e = 0; e < 8; ++e) acc[s][t][e] = 0.0f;

  const float* arow0 = A + (size_t)(m0 + lm) * lda;
  const float* arow1 = arow0 + (size_t)64 * lda;
  const float* bcol  = Bm + n0 + wnn;

  for (int k0 = 0; k0 < K; k0 += 32) {
    // stage B tile (permuted to fragment layout): 16 strided f32 loads -> one 32B LDS store
    {
      const float* bp = bcol + (size_t)(k0 + 16 * wkh) * ldb;
      bf16x16 wv;
#pragma unroll
      for (int e = 0; e < 16; ++e) wv[e] = f2bf(bp[(size_t)e * ldb]);
      *(bf16x16*)(&lbp[wt][wlane][0]) = wv;
      if (k0 + 32 < K) {
        __builtin_prefetch(bp + (size_t)32 * ldb, 0, 3);
        __builtin_prefetch(arow0 + k0 + 32, 0, 3);
        __builtin_prefetch(arow1 + k0 + 32, 0, 3);
      }
    }
    __syncthreads();

    bf16x16 af0 = load_afrag(arow0 + k0 + half * 16);
    bf16x16 af1 = load_afrag(arow1 + k0 + half * 16);

#pragma unroll
    for (int t = 0; t < 4; ++t) {
      bf16x16 bfr = *(const bf16x16*)(&lbp[t][lane][0]);
      acc[0][t] = wmma_bf16(af0, bfr, acc[0][t]);
      acc[1][t] = wmma_bf16(af1, bfr, acc[1][t]);
    }
    __syncthreads();
  }

#pragma unroll
  for (int s = 0; s < 2; ++s) {
#pragma unroll
    for (int t = 0; t < 4; ++t) {
#pragma unroll
      for (int r = 0; r < 8; ++r) {
        size_t idx = (size_t)(m0 + 64 * s + r + 8 * half) * ldc + (n0 + 16 * t + lm);
        float v = acc[s][t][r];
        if (EP == 1) v += Res[idx];
        else if (EP == 2) v = gelu_f(v);
        C[idx] = v;
      }
    }
  }
}

// ---------------- fused causal flash attention ----------------
// qkv layout: [B, T, 3*D], q at col h*64, k at D + h*64, v at 2D + h*64
// grid = (T/16, H, B), block = 32 (one wave per 16-query tile)
// V chunk is copied global->LDS with the CDNA5 async engine
// (global_load_async_to_lds_b128, ASYNCcnt), overlapped with the S=QK^T WMMAs
// and the online softmax; s_wait_asynccnt 0 gates the P@V fragment builds.
__global__ __launch_bounds__(32)
void attn_kernel(const float* __restrict__ qkv, float* __restrict__ out) {
  __shared__ alignas(16) float  vld[32][64];
  __shared__ __bf16 pld[16][32];
  const int lane = threadIdx.x;
  const int half = lane >> 4;
  const int lm   = lane & 15;
  const int qt = blockIdx.x, h = blockIdx.y, b = blockIdx.z;
  const int q0 = qt * 16;
  const int LDQ = 3 * DD;   // 2304
  const float* base = qkv + (size_t)b * TT * LDQ;

  // per-lane LDS destination (row = lane) for the async V copy
  unsigned vrow_lds = (unsigned)(uintptr_t)(&vld[lane][0]);

  // Q A-fragments for the two hd chunks (hd = 32c + half*16 + e)
  bf16x16 qa[2];
  {
    const float* qrow = base + (size_t)(q0 + lm) * LDQ + h * HDIM;
    qa[0] = load_afrag(qrow + half * 16);
    qa[1] = load_afrag(qrow + 32 + half * 16);
  }

  f32x8 acc[4];
#pragma unroll
  for (int t = 0; t < 4; ++t)
#pragma unroll
    for (int e = 0; e < 8; ++e) acc[t][e] = 0.0f;
  float mrow[8], lrow[8];
#pragma unroll
  for (int r = 0; r < 8; ++r) { mrow[r] = -1e30f; lrow[r] = 0.0f; }

  const float sscale = 0.125f;  // hd^-0.5 = 1/8

  for (int k0 = 0; k0 <= q0 + 15; k0 += 32) {
    // --- kick off async V copy (32 keys x 64 hd, one 256B row per lane) ---
    {
      // previous iteration's LDS fragment reads must be complete before the
      // async engine overwrites vld (async-to-LDS is not ordered with wave DS ops)
      asm volatile("s_wait_dscnt 0x0" ::: "memory");
      const float* vrow = base + (size_t)(k0 + lane) * LDQ + 2 * DD + h * HDIM;
#pragma unroll
      for (int f = 0; f < 16; ++f) {
        asm volatile("global_load_async_to_lds_b128 %0, %1, off offset:%2"
                     :: "v"(vrow_lds), "v"(vrow), "i"(16 * f) : "memory");
      }
    }

    // --- S = Q K^T for the two 16-key tiles (overlaps the async DMA) ---
    f32x8 stile[2];
#pragma unroll
    for (int t = 0; t < 2; ++t) {
      const float* krow = base + (size_t)(k0 + 16 * t + lm) * LDQ + DD + h * HDIM;
      bf16x16 kb0, kb1;
      {
        const float* p = krow + 16 * half;        // chunk c=0: hd = 16*half + e
        float4 y0 = *(const float4*)(p);
        float4 y1 = *(const float4*)(p + 4);
        float4 y2 = *(const float4*)(p + 8);
        float4 y3 = *(const float4*)(p + 12);
        kb0[0] = f2bf(y0.x); kb0[1] = f2bf(y0.y); kb0[2]  = f2bf(y0.z); kb0[3]  = f2bf(y0.w);
        kb0[4] = f2bf(y1.x); kb0[5] = f2bf(y1.y); kb0[6]  = f2bf(y1.z); kb0[7]  = f2bf(y1.w);
        kb0[8] = f2bf(y2.x); kb0[9] = f2bf(y2.y); kb0[10] = f2bf(y2.z); kb0[11] = f2bf(y2.w);
        kb0[12] = f2bf(y3.x); kb0[13] = f2bf(y3.y); kb0[14] = f2bf(y3.z); kb0[15] = f2bf(y3.w);
      }
      {
        const float* p = krow + 32 + 16 * half;   // chunk c=1
        float4 y0 = *(const float4*)(p);
        float4 y1 = *(const float4*)(p + 4);
        float4 y2 = *(const float4*)(p + 8);
        float4 y3 = *(const float4*)(p + 12);
        kb1[0] = f2bf(y0.x); kb1[1] = f2bf(y0.y); kb1[2]  = f2bf(y0.z); kb1[3]  = f2bf(y0.w);
        kb1[4] = f2bf(y1.x); kb1[5] = f2bf(y1.y); kb1[6]  = f2bf(y1.z); kb1[7]  = f2bf(y1.w);
        kb1[8] = f2bf(y2.x); kb1[9] = f2bf(y2.y); kb1[10] = f2bf(y2.z); kb1[11] = f2bf(y2.w);
        kb1[12] = f2bf(y3.x); kb1[13] = f2bf(y3.y); kb1[14] = f2bf(y3.z); kb1[15] = f2bf(y3.w);
      }
      f32x8 c;
#pragma unroll
      for (int e = 0; e < 8; ++e) c[e] = 0.0f;
      c = wmma_bf16(qa[0], kb0, c);
      c = wmma_bf16(qa[1], kb1, c);
      stile[t] = c;
    }

    // --- online softmax (rows r+8*half live in this lane's half-group) ---
    float pr0[8], pr1[8];
#pragma unroll
    for (int r = 0; r < 8; ++r) {
      int qrow_ = q0 + r + 8 * half;
      float s0 = stile[0][r] * sscale;
      float s1 = stile[1][r] * sscale;
      if (k0 + lm > qrow_)        s0 = -1e9f;
      if (k0 + 16 + lm > qrow_)   s1 = -1e9f;
      float mx = fmaxf(s0, s1);
      mx = fmaxf(mx, __shfl_xor(mx, 1, 32));
      mx = fmaxf(mx, __shfl_xor(mx, 2, 32));
      mx = fmaxf(mx, __shfl_xor(mx, 4, 32));
      mx = fmaxf(mx, __shfl_xor(mx, 8, 32));
      float mnew = fmaxf(mrow[r], mx);
      float sc = expf(mrow[r] - mnew);
      float p0 = expf(s0 - mnew);
      float p1 = expf(s1 - mnew);
      float sum = p0 + p1;
      sum += __shfl_xor(sum, 1, 32);
      sum += __shfl_xor(sum, 2, 32);
      sum += __shfl_xor(sum, 4, 32);
      sum += __shfl_xor(sum, 8, 32);
      lrow[r] = lrow[r] * sc + sum;
      mrow[r] = mnew;
#pragma unroll
      for (int t = 0; t < 4; ++t) acc[t][r] *= sc;
      pr0[r] = p0; pr1[r] = p1;
    }

    __syncthreads();
#pragma unroll
    for (int r = 0; r < 8; ++r) {
      pld[r + 8 * half][lm]      = f2bf(pr0[r]);
      pld[r + 8 * half][16 + lm] = f2bf(pr1[r]);
    }
    __syncthreads();

    // P A-fragment (contraction over 32 keys)
    bf16x16 pa;
#pragma unroll
    for (int e = 0; e < 16; ++e) {
      int kidx = half * 16 + e;
      pa[e] = pld[lm][kidx];
    }

    // wait for the async V copy, then O += P @ V  (four 16-wide hd tiles)
    asm volatile("s_wait_asynccnt 0x0" ::: "memory");
    __syncthreads();
#pragma unroll
    for (int t = 0; t < 4; ++t) {
      bf16x16 vb;
#pragma unroll
      for (int e = 0; e < 16; ++e) vb[e] = f2bf(vld[16 * half + e][16 * t + lm]);
      acc[t] = wmma_bf16(pa, vb, acc[t]);
    }
  }

  // write normalized output into [B,T,D]
#pragma unroll
  for (int r = 0; r < 8; ++r) {
    int row = q0 + r + 8 * half;
    float inv = 1.0f / lrow[r];
#pragma unroll
    for (int t = 0; t < 4; ++t)
      out[((size_t)(b * TT + row)) * DD + h * HDIM + 16 * t + lm] = acc[t][r] * inv;
  }
}

// ---------------- host orchestration ----------------
extern "C" void kernel_launch(void* const* d_in, const int* in_sizes, int n_in,
                              void* d_out, int out_size, void* d_ws, size_t ws_size,
                              hipStream_t stream) {
  (void)in_sizes; (void)n_in; (void)out_size; (void)ws_size;
  const int*   ids     = (const int*)d_in[0];
  const float* tok_emb = (const float*)d_in[1];
  const float* pos_emb = (const float*)d_in[2];
  const float* ln1_g   = (const float*)d_in[3];
  const float* ln1_b   = (const float*)d_in[4];
  const float* Wqkv    = (const float*)d_in[5];
  const float* Wo      = (const float*)d_in[6];
  const float* ln2_g   = (const float*)d_in[7];
  const float* ln2_b   = (const float*)d_in[8];
  const float* W1      = (const float*)d_in[9];
  const float* W2      = (const float*)d_in[10];
  const float* pW1     = (const float*)d_in[11];
  const float* pW2     = (const float*)d_in[12];
  const float* lnf_g   = (const float*)d_in[13];
  const float* lnf_b   = (const float*)d_in[14];
  const float* W_lm    = (const float*)d_in[15];
  float* out = (float*)d_out;

  float* p = (float*)d_ws;
  float* X    = p; p += (size_t)BT * DD;
  float* Hb   = p; p += (size_t)BT * DD;
  float* QKV  = p; p += (size_t)BT * 3 * DD;
  float* ATTN = p; p += (size_t)BT * DD;    // also reused as V-projection buffer
  float* FFb  = p; p += (size_t)BT * FFD;
  float* R1   = p; p += (size_t)BT * DD;
  float* AD   = p; p += (size_t)BT * DD;
  float* AH   = p; p += (size_t)BT * PHD;
  float* ED   = p; p += BT;
  float* THq  = p; p += 4;

  const int EW_GRID = (BT * DD + 255) / 256;

  embed_kernel<<<EW_GRID, 256, 0, stream>>>(ids, tok_emb, pos_emb, X);

  for (int i = 0; i < LL; ++i) {
    // ---- full block ----
    ln_kernel<<<BT, 256, 0, stream>>>(X, ln1_g + i * DD, ln1_b + i * DD, Hb);
    gemm_wmma<0><<<dim3(3 * DD / 64, BT / 128), 128, 0, stream>>>(
        Hb, Wqkv + (size_t)i * DD * 3 * DD, nullptr, QKV, DD, DD, 3 * DD, 3 * DD);
    attn_kernel<<<dim3(TT / 16, HH, BB), 32, 0, stream>>>(QKV, ATTN);
    gemm_wmma<1><<<dim3(DD / 64, BT / 128), 128, 0, stream>>>(
        ATTN, Wo + (size_t)i * DD * DD, X, X, DD, DD, DD, DD);
    ln_kernel<<<BT, 256, 0, stream>>>(X, ln2_g + i * DD, ln2_b + i * DD, Hb);
    gemm_wmma<2><<<dim3(FFD / 64, BT / 128), 128, 0, stream>>>(
        Hb, W1 + (size_t)i * DD * FFD, nullptr, FFb, DD, DD, FFD, FFD);
    gemm_wmma<1><<<dim3(DD / 64, BT / 128), 128, 0, stream>>>(
        FFb, W2 + (size_t)i * FFD * DD, X, X, FFD, FFD, DD, DD);

    // ---- routing statistics ----
    rowmeansq_kernel<<<BT, 256, 0, stream>>>(X, ED);
    quantile_kernel<<<1, 256, 0, stream>>>(ED, THq);

    // ---- reprocess path (route == 2) using previous layer's weights ----
    if (i > 0) {
      int j = i - 1;
      ln_kernel<<<BT, 256, 0, stream>>>(X, ln1_g + j * DD, ln1_b + j * DD, Hb);
      // v = h @ Wqkv[j][:, 2D:3D]  (column-slice: pointer offset 2D, ldb = 3D)
      gemm_wmma<0><<<dim3(DD / 64, BT / 128), 128, 0, stream>>>(
          Hb, Wqkv + (size_t)j * DD * 3 * DD + 2 * DD, nullptr, ATTN, DD, DD, 3 * DD, DD);
      gemm_wmma<1><<<dim3(DD / 64, BT / 128), 128, 0, stream>>>(
          ATTN, Wo + (size_t)j * DD * DD, X, R1, DD, DD, DD, DD);
      ln_kernel<<<BT, 256, 0, stream>>>(R1, ln2_g + j * DD, ln2_b + j * DD, Hb);
      gemm_wmma<2><<<dim3(FFD / 64, BT / 128), 128, 0, stream>>>(
          Hb, W1 + (size_t)j * DD * FFD, nullptr, FFb, DD, DD, FFD, FFD);
      gemm_wmma<1><<<dim3(DD / 64, BT / 128), 128, 0, stream>>>(
          FFb, W2 + (size_t)j * FFD * DD, R1, R1, FFD, FFD, DD, DD);
      select_kernel<<<EW_GRID, 256, 0, stream>>>(X, R1, ED, THq, 2);
    }

    // ---- plastic adapter path (route == 3) ----
    gemm_wmma<2><<<dim3(PHD / 64, BT / 128), 128, 0, stream>>>(
        X, pW1, nullptr, AH, DD, DD, PHD, PHD);
    gemm_wmma<1><<<dim3(DD / 64, BT / 128), 128, 0, stream>>>(
        AH, pW2, X, AD, PHD, PHD, DD, DD);
    select_kernel<<<EW_GRID, 256, 0, stream>>>(X, AD, ED, THq, 3);
  }

  // ---- final layernorm + LM head ----
  ln_kernel<<<BT, 256, 0, stream>>>(X, lnf_g, lnf_b, Hb);
  gemm_wmma<0><<<dim3(VV / 64, BT / 128), 128, 0, stream>>>(
      Hb, W_lm, nullptr, out, DD, DD, VV, VV);
}
